// TextGeneratorModel_67104569033380
// MI455X (gfx1250) — matrix operand order repaired and checked by reading, not compiled, not run on hardware
//
#include <hip/hip_runtime.h>
#include <math.h>

// ---- model dims (fixed by reference) ----
#define LL   6
#define HHN  16
#define EE   1024
#define DKK  64
#define FFF  4096
#define VV   32000
#define BBN  2
#define TTN  1024
#define MTOT (BBN * TTN)   // 2048
#define HDK  (HHN * DKK)   // 1024

typedef __attribute__((ext_vector_type(16))) __bf16 v16bf;
typedef __attribute__((ext_vector_type(8)))  float  v8f;
typedef __attribute__((ext_vector_type(4)))  unsigned int u32x4;
typedef __attribute__((ext_vector_type(4)))  float        f32x4;

struct __align__(16) Raw32 { u32x4 lo, hi; };

__device__ __forceinline__ unsigned short f2bf(float f) {
  union { float f; unsigned int u; } c; c.f = f;
  unsigned int u = c.u;
  return (unsigned short)((u + 0x7FFFu + ((u >> 16) & 1u)) >> 16);
}

// A-fragment (16x32 bf16, M x K): lane m = lane%16, K chunks {kb..kb+7} and {16+kb..16+kb+7}, kb = 8*(lane/16)
__device__ __forceinline__ v16bf lds_afrag(const unsigned short* tile, int ldk, int m0, int kk) {
  int lane = threadIdx.x & 31;
  const unsigned short* p = tile + (m0 + (lane & 15)) * ldk + kk + ((lane >> 4) << 3);
  Raw32 r;
  r.lo = *reinterpret_cast<const u32x4*>(p);
  r.hi = *reinterpret_cast<const u32x4*>(p + 16);
  return __builtin_bit_cast(v16bf, r);
}

// B-fragment from K-transposed LDS tile Bt[N][K]: lane n = lane%16, K chunks {kb..kb+7} and {16+kb..16+kb+7}, kb = 8*(lane/16)
__device__ __forceinline__ v16bf lds_bfrag(const unsigned short* bt, int ldk, int n0, int kk) {
  int lane = threadIdx.x & 31;
  const unsigned short* p = bt + (n0 + (lane & 15)) * ldk + kk + ((lane >> 4) << 3);
  Raw32 r;
  r.lo = *reinterpret_cast<const u32x4*>(p);
  r.hi = *reinterpret_cast<const u32x4*>(p + 16);
  return __builtin_bit_cast(v16bf, r);
}

// ---------------- generic GEMM: C[M,N] = A_bf16[M,K] * B_f32[K,N] (+bias,+residual,ReLU) ----------------
// Tiles: BM=256 (8 waves x 32 rows), BN=64, BK=32, double-buffered LDS.
// bperm==1: B is Wq/Wk/Wv with layout [H][E][DK]; logical B[k][n] = src[(n>>6)*K*64 + k*64 + (n&63)]
__global__ __launch_bounds__(256) void gemm_bf16_kernel(
    const unsigned short* __restrict__ A, const float* __restrict__ Bsrc,
    const float* __restrict__ bias, const float* __restrict__ resid,
    float* __restrict__ outf, unsigned short* __restrict__ outb,
    int M, int N, int K, int relu, int bperm)
{
  __shared__ alignas(16) unsigned short As[2][256 * 32];
  __shared__ alignas(16) unsigned short Bt[2][64 * 32];
  int tid = threadIdx.x;
  int wave = tid >> 5, lane = tid & 31;
  int m0 = blockIdx.y * 256, n0 = blockIdx.x * 64;

  v8f acc[2][4];
  #pragma unroll
  for (int mt = 0; mt < 2; ++mt)
    #pragma unroll
    for (int nt = 0; nt < 4; ++nt)
      #pragma unroll
      for (int j = 0; j < 8; ++j) acc[mt][nt][j] = 0.f;

  // staging mapping: A -> thread owns one full 32-half row; B -> 8 floats/thread
  int bk = tid >> 3, bn = (tid & 7) * 8;

  u32x4 areg[4];
  f32x4 breg[2];

  auto load_tile = [&](int ks) {
    const unsigned short* ga = A + (size_t)(m0 + tid) * K + ks * 32;
    #pragma unroll
    for (int i = 0; i < 4; ++i) areg[i] = *reinterpret_cast<const u32x4*>(ga + i * 8);
    int k0 = ks * 32;
    const float* gb;
    if (!bperm) {
      gb = Bsrc + (size_t)(k0 + bk) * N + n0 + bn;
    } else {
      int n = n0 + bn;
      gb = Bsrc + (size_t)(n >> 6) * K * 64 + (size_t)(k0 + bk) * 64 + (n & 63);
    }
    breg[0] = *reinterpret_cast<const f32x4*>(gb);
    breg[1] = *reinterpret_cast<const f32x4*>(gb + 4);
  };

  auto store_tile = [&](int buf) {
    unsigned short* as = As[buf];
    #pragma unroll
    for (int i = 0; i < 4; ++i)
      *reinterpret_cast<u32x4*>(&as[tid * 32 + i * 8]) = areg[i];
    unsigned short* bt = Bt[buf];
    #pragma unroll
    for (int i = 0; i < 4; ++i) bt[(bn + i) * 32 + bk]     = f2bf(breg[0][i]);
    #pragma unroll
    for (int i = 0; i < 4; ++i) bt[(bn + 4 + i) * 32 + bk] = f2bf(breg[1][i]);
  };

  auto compute = [&](int buf) {
    const unsigned short* as = As[buf];
    const unsigned short* bt = Bt[buf];
    v16bf a0 = lds_afrag(as, 32, wave * 32, 0);
    v16bf a1 = lds_afrag(as, 32, wave * 32 + 16, 0);
    v16bf b[4];
    #pragma unroll
    for (int nt = 0; nt < 4; ++nt) b[nt] = lds_bfrag(bt, 32, nt * 16, 0);
    #pragma unroll
    for (int nt = 0; nt < 4; ++nt) {
      acc[0][nt] = __builtin_amdgcn_wmma_f32_16x16x32_bf16(false, a0, false, b[nt], (short)0, acc[0][nt], false, false);
      acc[1][nt] = __builtin_amdgcn_wmma_f32_16x16x32_bf16(false, a1, false, b[nt], (short)0, acc[1][nt], false, false);
    }
  };

  int nk = K >> 5;
  load_tile(0);
  store_tile(0);
  for (int ks = 0; ks < nk; ++ks) {
    __syncthreads();
    bool more = (ks + 1) < nk;
    if (more) load_tile(ks + 1);          // global loads in flight during compute
    compute(ks & 1);
    if (more) store_tile((ks + 1) & 1);   // waits on loads only here
  }

  int ncol = n0 + (lane & 15);
  #pragma unroll
  for (int mt = 0; mt < 2; ++mt) {
    int mrow = m0 + wave * 32 + mt * 16 + ((lane >> 4) << 3);
    #pragma unroll
    for (int nt = 0; nt < 4; ++nt) {
      int n = ncol + nt * 16;
      float bsv = bias ? bias[n] : 0.f;
      #pragma unroll
      for (int j = 0; j < 8; ++j) {
        int m = mrow + j;
        float vv = acc[mt][nt][j] + bsv;
        if (resid) vv += resid[(size_t)m * N + n];
        if (relu)  vv = fmaxf(vv, 0.f);
        if (outf)  outf[(size_t)m * N + n] = vv;
        if (outb)  outb[(size_t)m * N + n] = f2bf(vv);
      }
    }
  }
}

// ---------------- flash attention: per (b,h), 64-row q tile, online softmax ----------------
__global__ __launch_bounds__(128) void attn_kernel(
    const unsigned short* __restrict__ q, const unsigned short* __restrict__ k,
    const unsigned short* __restrict__ v, unsigned short* __restrict__ o)
{
  __shared__ alignas(16) unsigned short Qs[64 * 64];
  __shared__ alignas(16) unsigned short Ks[64 * 64];   // [s][d] == Bt layout for S = Q*K^T
  __shared__ alignas(16) unsigned short Vt[64 * 64];   // [d][s] == Bt layout for O = P*V
  __shared__ alignas(16) unsigned short Ps[64 * 64];

  int tid = threadIdx.x;
  int wave = tid >> 5, lane = tid & 31;
  int hi = lane >> 4, nlo = lane & 15;
  int b = blockIdx.y >> 4, h = blockIdx.y & 15;
  int t0 = blockIdx.x * 64;

  int lr = tid >> 1;          // 0..63
  int lc = (tid & 1) * 32;    // 0 or 32

  { // Q tile (row-major [t][d])
    const unsigned short* g = q + (size_t)(b * TTN + t0 + lr) * HDK + h * DKK + lc;
    #pragma unroll
    for (int u = 0; u < 4; ++u)
      *reinterpret_cast<u32x4*>(&Qs[lr * 64 + lc + u * 8]) = *reinterpret_cast<const u32x4*>(g + u * 8);
  }

  float mst[8], lst[8];
  #pragma unroll
  for (int j = 0; j < 8; ++j) { mst[j] = -1e30f; lst[j] = 0.f; }
  v8f oacc[4];
  #pragma unroll
  for (int nt = 0; nt < 4; ++nt)
    #pragma unroll
    for (int j = 0; j < 8; ++j) oacc[nt][j] = 0.f;

  for (int st = 0; st <= (int)blockIdx.x; ++st) {
    int s0 = st * 64;
    __syncthreads();
    { // K tile direct; V tile transposed
      const unsigned short* gk = k + (size_t)(b * TTN + s0 + lr) * HDK + h * DKK + lc;
      #pragma unroll
      for (int u = 0; u < 4; ++u)
        *reinterpret_cast<u32x4*>(&Ks[lr * 64 + lc + u * 8]) = *reinterpret_cast<const u32x4*>(gk + u * 8);
      const unsigned short* gv = v + (size_t)(b * TTN + s0 + lr) * HDK + h * DKK + lc;
      #pragma unroll
      for (int u = 0; u < 4; ++u) {
        union { u32x4 q4; unsigned short s[8]; } un;
        un.q4 = *reinterpret_cast<const u32x4*>(gv + u * 8);
        int d = lc + u * 8;
        #pragma unroll
        for (int i = 0; i < 8; ++i) Vt[(d + i) * 64 + lr] = un.s[i];
      }
    }
    __syncthreads();

    // S = Q * K^T   (K-dim = 64 -> two WMMA steps; preload 4 B frags per step)
    v8f sacc[4];
    #pragma unroll
    for (int nt = 0; nt < 4; ++nt)
      #pragma unroll
      for (int j = 0; j < 8; ++j) sacc[nt][j] = 0.f;
    #pragma unroll
    for (int kk = 0; kk < 64; kk += 32) {
      v16bf a = lds_afrag(Qs, 64, wave * 16, kk);
      v16bf bb[4];
      #pragma unroll
      for (int nt = 0; nt < 4; ++nt) bb[nt] = lds_bfrag(Ks, 64, nt * 16, kk);
      #pragma unroll
      for (int nt = 0; nt < 4; ++nt)
        sacc[nt] = __builtin_amdgcn_wmma_f32_16x16x32_bf16(false, a, false, bb[nt], (short)0, sacc[nt], false, false);
    }

    // scale (1/sqrt(E) = 1/32) + causal mask
    #pragma unroll
    for (int nt = 0; nt < 4; ++nt) {
      int s = s0 + nt * 16 + nlo;
      #pragma unroll
      for (int j = 0; j < 8; ++j) {
        int t = t0 + wave * 16 + (hi << 3) + j;
        float sv = sacc[nt][j] * 0.03125f;
        sacc[nt][j] = (s > t) ? -1e30f : sv;
      }
    }

    // online softmax: row stats reduced across the 16-lane half holding each row set
    float rsum[8], corr[8];
    #pragma unroll
    for (int j = 0; j < 8; ++j) {
      float mx = fmaxf(fmaxf(sacc[0][j], sacc[1][j]), fmaxf(sacc[2][j], sacc[3][j]));
      #pragma unroll
      for (int msk = 1; msk <= 8; msk <<= 1)
        mx = fmaxf(mx, __shfl_xor(mx, msk, 32));
      float mnew = fmaxf(mst[j], mx);
      corr[j] = expf(mst[j] - mnew);
      mst[j] = mnew;
      rsum[j] = 0.f;
    }
    #pragma unroll
    for (int nt = 0; nt < 4; ++nt)
      #pragma unroll
      for (int j = 0; j < 8; ++j) {
        float p = expf(sacc[nt][j] - mst[j]);
        sacc[nt][j] = p;
        rsum[j] += p;
      }
    #pragma unroll
    for (int j = 0; j < 8; ++j) {
      #pragma unroll
      for (int msk = 1; msk <= 8; msk <<= 1)
        rsum[j] += __shfl_xor(rsum[j], msk, 32);
      lst[j] = lst[j] * corr[j] + rsum[j];
    }
    #pragma unroll
    for (int nt = 0; nt < 4; ++nt)
      #pragma unroll
      for (int j = 0; j < 8; ++j) oacc[nt][j] *= corr[j];

    // stage P via LDS (register C-layout -> A-fragment layout)
    #pragma unroll
    for (int nt = 0; nt < 4; ++nt)
      #pragma unroll
      for (int j = 0; j < 8; ++j)
        Ps[(wave * 16 + (hi << 3) + j) * 64 + nt * 16 + nlo] = f2bf(sacc[nt][j]);
    __syncthreads();

    // O += P * V
    #pragma unroll
    for (int kk = 0; kk < 64; kk += 32) {
      v16bf a = lds_afrag(Ps, 64, wave * 16, kk);
      v16bf bb[4];
      #pragma unroll
      for (int nt = 0; nt < 4; ++nt) bb[nt] = lds_bfrag(Vt, 64, nt * 16, kk);
      #pragma unroll
      for (int nt = 0; nt < 4; ++nt)
        oacc[nt] = __builtin_amdgcn_wmma_f32_16x16x32_bf16(false, a, false, bb[nt], (short)0, oacc[nt], false, false);
    }
  }

  // normalize + store bf16 at [b, t, h, d]
  #pragma unroll
  for (int nt = 0; nt < 4; ++nt) {
    int d = nt * 16 + nlo;
    #pragma unroll
    for (int j = 0; j < 8; ++j) {
      int t = t0 + wave * 16 + (hi << 3) + j;
      o[(size_t)(b * TTN + t) * HDK + h * DKK + d] = f2bf(oacc[nt][j] / lst[j]);
    }
  }
}

// ---------------- LayerNorm (fp32 in -> bf16 out) ----------------
__global__ __launch_bounds__(256) void ln_kernel(
    const float* __restrict__ x, const float* __restrict__ g, const float* __restrict__ bta,
    unsigned short* __restrict__ out)
{
  __shared__ float red[256];
  int row = blockIdx.x, tid = threadIdx.x;
  const float* xr = x + (size_t)row * EE;
  float v0[4];
  #pragma unroll
  for (int u = 0; u < 4; ++u) v0[u] = xr[u * 256 + tid];
  red[tid] = v0[0] + v0[1] + v0[2] + v0[3];
  __syncthreads();
  for (int off = 128; off > 0; off >>= 1) { if (tid < off) red[tid] += red[tid + off]; __syncthreads(); }
  float mu = red[0] * (1.f / EE);
  __syncthreads();
  float ss = 0.f;
  #pragma unroll
  for (int u = 0; u < 4; ++u) { float d = v0[u] - mu; ss += d * d; }
  red[tid] = ss;
  __syncthreads();
  for (int off = 128; off > 0; off >>= 1) { if (tid < off) red[tid] += red[tid + off]; __syncthreads(); }
  float rstd = rsqrtf(red[0] * (1.f / EE) + 1e-5f);
  #pragma unroll
  for (int u = 0; u < 4; ++u) {
    int i = u * 256 + tid;
    out[(size_t)row * EE + i] = f2bf((v0[u] - mu) * rstd * g[i] + bta[i]);
  }
}

// ---------------- embedding + positional ----------------
__global__ __launch_bounds__(256) void embed_kernel(
    const int* __restrict__ idx, const float* __restrict__ emb, const float* __restrict__ pos,
    float* __restrict__ x)
{
  int row = blockIdx.x, tid = threadIdx.x;
  int tok = idx[row];
  int t = row % TTN;
  #pragma unroll
  for (int u = 0; u < 4; ++u) {
    int i = u * 256 + tid;
    x[(size_t)row * EE + i] = emb[(size_t)tok * EE + i] + pos[(size_t)t * EE + i];
  }
}

extern "C" void kernel_launch(void* const* d_in, const int* in_sizes, int n_in,
                              void* d_out, int out_size, void* d_ws, size_t ws_size,
                              hipStream_t stream) {
  (void)in_sizes; (void)n_in; (void)out_size; (void)ws_size;
  const int*   idx  = (const int*)  d_in[0];
  const float* emb  = (const float*)d_in[1];
  const float* pos  = (const float*)d_in[2];
  const float* Wq   = (const float*)d_in[3];
  const float* Wk   = (const float*)d_in[4];
  const float* Wv   = (const float*)d_in[5];
  const float* Wp   = (const float*)d_in[6];
  const float* bp   = (const float*)d_in[7];
  const float* ln1g = (const float*)d_in[8];
  const float* ln1b = (const float*)d_in[9];
  const float* ln2g = (const float*)d_in[10];
  const float* ln2b = (const float*)d_in[11];
  const float* W1   = (const float*)d_in[12];
  const float* b1   = (const float*)d_in[13];
  const float* W2   = (const float*)d_in[14];
  const float* b2   = (const float*)d_in[15];
  const float* lnfg = (const float*)d_in[16];
  const float* lnfb = (const float*)d_in[17];
  const float* Wout = (const float*)d_in[18];
  const float* bout = (const float*)d_in[19];
  float* out = (float*)d_out;

  char* w = (char*)d_ws;
  float*          x    = (float*)w;          w += (size_t)MTOT * EE * 4;
  unsigned short* h_bf = (unsigned short*)w; w += (size_t)MTOT * EE * 2;
  unsigned short* q_bf = (unsigned short*)w; w += (size_t)MTOT * HDK * 2;
  unsigned short* k_bf = (unsigned short*)w; w += (size_t)MTOT * HDK * 2;
  unsigned short* v_bf = (unsigned short*)w; w += (size_t)MTOT * HDK * 2;
  unsigned short* o_bf = (unsigned short*)w; w += (size_t)MTOT * HDK * 2;
  unsigned short* f1   = (unsigned short*)w; w += (size_t)MTOT * FFF * 2;

  auto gemm = [&](const unsigned short* A, const float* Bm, const float* bias,
                  const float* res, float* of, unsigned short* ob,
                  int M, int N, int K, int relu, int perm) {
    dim3 grid(N / 64, M / 256);
    gemm_bf16_kernel<<<grid, 256, 0, stream>>>(A, Bm, bias, res, of, ob, M, N, K, relu, perm);
  };

  embed_kernel<<<MTOT, 256, 0, stream>>>(idx, emb, pos, x);

  for (int l = 0; l < LL; ++l) {
    const float* wq = Wq + (size_t)l * HHN * EE * DKK;
    const float* wk = Wk + (size_t)l * HHN * EE * DKK;
    const float* wv = Wv + (size_t)l * HHN * EE * DKK;

    ln_kernel<<<MTOT, 256, 0, stream>>>(x, ln1g + (size_t)l * EE, ln1b + (size_t)l * EE, h_bf);
    gemm(h_bf, wq, nullptr, nullptr, nullptr, q_bf, MTOT, HDK, EE, 0, 1);
    gemm(h_bf, wk, nullptr, nullptr, nullptr, k_bf, MTOT, HDK, EE, 0, 1);
    gemm(h_bf, wv, nullptr, nullptr, nullptr, v_bf, MTOT, HDK, EE, 0, 1);
    attn_kernel<<<dim3(TTN / 64, BBN * HHN), 128, 0, stream>>>(q_bf, k_bf, v_bf, o_bf);
    gemm(o_bf, Wp + (size_t)l * EE * EE, bp + (size_t)l * EE, x, x, nullptr, MTOT, EE, EE, 0, 0);

    ln_kernel<<<MTOT, 256, 0, stream>>>(x, ln2g + (size_t)l * EE, ln2b + (size_t)l * EE, h_bf);
    gemm(h_bf, W1 + (size_t)l * EE * FFF, b1 + (size_t)l * FFF, nullptr, nullptr, f1, MTOT, FFF, EE, 1, 0);
    gemm(f1, W2 + (size_t)l * FFF * EE, b2 + (size_t)l * EE, x, x, nullptr, MTOT, EE, FFF, 0, 0);
  }

  ln_kernel<<<MTOT, 256, 0, stream>>>(x, lnfg, lnfb, h_bf);
  gemm(h_bf, Wout, bout, nullptr, out, nullptr, MTOT, VV, EE, 0, 0);
}
